// MidBlock_644245095147
// MI455X (gfx1250) — matrix operand order repaired and checked by reading, not compiled, not run on hardware
//
#include <hip/hip_runtime.h>
#include <hip/hip_bf16.h>
#include <math.h>

// Problem constants (reference: B=16, C=512, H=W=32, GROUPS=32)
#define CCH 512
#define BB  16
#define HH  32
#define WWD 32
#define HWP 1024          // H*W
#define CPG 16            // channels per group (512/32)

typedef __attribute__((ext_vector_type(16))) __bf16 v16bf;
typedef __attribute__((ext_vector_type(8)))  float  v8f;
typedef int v4i_ __attribute__((__vector_size__(4 * sizeof(int))));

union Frag16 { v16bf v; unsigned u[8]; };

// ---------------------------------------------------------------------------
// CDNA5 async global->LDS copy (ASYNCcnt path), with sync fallback.
// ---------------------------------------------------------------------------
#if defined(__has_builtin)
#  if __has_builtin(__builtin_amdgcn_global_load_async_to_lds_b128) && \
      __has_builtin(__builtin_amdgcn_s_wait_asynccnt)
#    define USE_ASYNC_LDS 1
#  endif
#endif
#ifndef USE_ASYNC_LDS
#  define USE_ASYNC_LDS 0
#endif

// Copy 32 bytes (per lane) global -> LDS.
__device__ __forceinline__ void cp32(void* lds, const void* gsrc) {
#if USE_ASYNC_LDS
  __attribute__((address_space(1))) v4i_* g =
      (__attribute__((address_space(1))) v4i_*)const_cast<void*>(gsrc);
  __attribute__((address_space(3))) v4i_* l =
      (__attribute__((address_space(3))) v4i_*)lds;
  __builtin_amdgcn_global_load_async_to_lds_b128(g, l, 0, 0);
  __builtin_amdgcn_global_load_async_to_lds_b128(g, l, 16, 0);
#else
  const uint4* s = (const uint4*)gsrc;
  uint4 a0 = s[0], a1 = s[1];
  ((uint4*)lds)[0] = a0;
  ((uint4*)lds)[1] = a1;
#endif
}
// Wait: <=4 async ops outstanding (the prefetched next step), or all done.
__device__ __forceinline__ void wait_async_le4() {
#if USE_ASYNC_LDS
  __builtin_amdgcn_s_wait_asynccnt(4);
#endif
}
__device__ __forceinline__ void wait_async_le0() {
#if USE_ASYNC_LDS
  __builtin_amdgcn_s_wait_asynccnt(0);
#endif
}

// ---------------------------------------------------------------------------
// Layout prep kernels: fp32 -> bf16 in GEMM-friendly layouts
// ---------------------------------------------------------------------------

// conv3x3 weights [O][I][3][3] f32 -> [tap][O][I] bf16 (K ordered tap-major)
__global__ void k_w3_bf16(const float* __restrict__ w, __bf16* __restrict__ out) {
  int idx = blockIdx.x * 256 + threadIdx.x;
  if (idx >= CCH * CCH * 9) return;
  const int i = idx & 511;
  const int r = idx >> 9;
  const int o = r & 511;
  const int t = r >> 9;                       // tap = ky*3+kx
  out[idx] = (__bf16)w[((size_t)o * CCH + i) * 9 + t];
}

// 1x1 weights [O][I] f32 -> bf16 (single tap)
__global__ void k_w1_bf16(const float* __restrict__ w, __bf16* __restrict__ out) {
  int idx = blockIdx.x * 256 + threadIdx.x;
  if (idx >= CCH * CCH) return;
  out[idx] = (__bf16)w[idx];
}

// x [B][C][H][W] f32 -> pixel-major bf16 [B][HW][C]
__global__ void k_x_pm(const float* __restrict__ x, __bf16* __restrict__ out) {
  int idx = blockIdx.x * 256 + threadIdx.x;
  if (idx >= BB * CCH * HWP) return;
  const int p = idx & 1023;
  const int r = idx >> 10;
  const int c = r & 511;
  const int b = r >> 9;
  out[((size_t)b * HWP + p) * CCH + c] = (__bf16)x[idx];
}

// ---------------------------------------------------------------------------
// Implicit-GEMM convolution (ntaps = 9 for 3x3 pad=1, ntaps = 1 for 1x1).
// 128x128 tile / 8-wave workgroup; double-buffered LDS with async prefetch.
// ---------------------------------------------------------------------------
__global__ __launch_bounds__(256)
void conv_gemm(const __bf16* __restrict__ Xpm,   // [B][HW][C]
               const __bf16* __restrict__ Wt,    // [ntaps][C][C]
               const float*  __restrict__ bias,  // [C] or null
               int ntaps, float out_scale,
               const float*  __restrict__ resid,   // f32 NCHW or null
               float*        __restrict__ out_f32, // f32 NCHW or null
               __bf16*       __restrict__ out_pm,  // bf16 [B][HW][C] or null
               __bf16*       __restrict__ out_cm)  // bf16 [B][C][HW] or null
{
  __shared__ __align__(16) __bf16 Asub[2][128][32];   // [buf][m_local][k]
  __shared__ __align__(16) __bf16 Bsub[2][128][32];   // [buf][n_local][k]

  const int b    = blockIdx.z;
  const int m0   = blockIdx.y * 128;
  const int n0   = blockIdx.x * 128;
  const int tid  = threadIdx.x;
  const int wv   = tid >> 5;
  const int lane = tid & 31;
  const int hf   = lane >> 4;
  const int lm   = lane & 15;

  const v8f vzero = {0.f,0.f,0.f,0.f,0.f,0.f,0.f,0.f};
  v8f acc[8];
#pragma unroll
  for (int i = 0; i < 8; ++i) acc[i] = vzero;

  const int srow = tid >> 1;          // staging row 0..127
  const int kh   = (tid & 1) * 16;    // k half: 0 or 16
  const int nsteps = ntaps * 16;      // 32-wide k-steps

  // Issue 4 async copy instructions per wave (uniform, unconditional).
  auto stage = [&](int step, int buf) {
    const int tap = step >> 4;
    const int kc  = (step & 15) * 32;
    const int dy = (ntaps == 1) ? 0 : (tap / 3 - 1);
    const int dx = (ntaps == 1) ? 0 : (tap % 3 - 1);
    cp32(&Asub[buf][srow][kh],
         Wt + ((size_t)tap * CCH + (m0 + srow)) * CCH + kc + kh);
    const int p = n0 + srow;
    int py = (p >> 5) + dy;
    int px = (p & 31) + dx;
    py = py < 0 ? 0 : (py > HH - 1 ? HH - 1 : py);    // clamp; OOB zero-fixed later
    px = px < 0 ? 0 : (px > WWD - 1 ? WWD - 1 : px);
    cp32(&Bsub[buf][srow][kh],
         Xpm + ((size_t)b * HWP + (py * WWD + px)) * CCH + kc + kh);
  };
  auto oob = [&](int step) -> bool {
    const int tap = step >> 4;
    const int dy = (ntaps == 1) ? 0 : (tap / 3 - 1);
    const int dx = (ntaps == 1) ? 0 : (tap % 3 - 1);
    const int p  = n0 + srow;
    const int py = (p >> 5) + dy;
    const int px = (p & 31) + dx;
    return (py < 0) || (py >= HH) || (px < 0) || (px >= WWD);
  };

  stage(0, 0);
  for (int s = 0; s < nsteps; ++s) {
    const int cur = s & 1;
    if (s + 1 < nsteps) { stage(s + 1, cur ^ 1); wait_async_le4(); }
    else                { wait_async_le0(); }
    if (ntaps != 1 && oob(s)) {          // zero-fix padded pixels in this buffer
      const uint4 z = make_uint4(0u,0u,0u,0u);
      *reinterpret_cast<uint4*>(&Bsub[cur][srow][kh])     = z;
      *reinterpret_cast<uint4*>(&Bsub[cur][srow][kh + 8]) = z;
    }
    __syncthreads();

    // A fragment (16x32, contiguous k = hf*16 + s, matching B)
    Frag16 af;
    const int arow = wv * 16 + lm;
#pragma unroll
    for (int v = 0; v < 8; ++v) {
      const int kb = hf * 16 + v * 2;
      af.u[v] = *reinterpret_cast<const unsigned*>(&Asub[cur][arow][kb]);
    }
#pragma unroll
    for (int nt = 0; nt < 8; ++nt) {
      Frag16 bfr;
      const int brow = nt * 16 + lm;
#pragma unroll
      for (int v = 0; v < 8; ++v)
        bfr.u[v] = *reinterpret_cast<const unsigned*>(&Bsub[cur][brow][hf * 16 + v * 2]);
      acc[nt] = __builtin_amdgcn_wmma_f32_16x16x32_bf16(
          false, af.v, false, bfr.v, (short)0, acc[nt], false, false);
    }
    __syncthreads();
  }

  // Epilogue: D layout lane(hf,lm), VGPR v -> m = hf*8+v, n = lm
#pragma unroll
  for (int nt = 0; nt < 8; ++nt) {
    const int n = n0 + nt * 16 + lm;
#pragma unroll
    for (int v = 0; v < 8; ++v) {
      const int m = m0 + wv * 16 + hf * 8 + v;
      float val = acc[nt][v];
      if (bias) val += bias[m];
      val *= out_scale;
      const size_t onchw = ((size_t)b * CCH + m) * HWP + n;
      if (resid)   val += resid[onchw];
      if (out_f32) out_f32[onchw] = val;
      if (out_pm)  out_pm[((size_t)b * HWP + n) * CCH + m] = (__bf16)val;
      if (out_cm)  out_cm[onchw] = (__bf16)val;
    }
  }
}

// ---------------------------------------------------------------------------
// Batched attention GEMM (both operands row-major in K):
//   C[b][row][col] = sum_k A[b][row][k] * Bt[b][col][k]
// A: [B][1024][Kdim], Bt: [B][Ncols][Kdim]. Scores: Kdim=512, Ncols=1024.
// P@V: A=P [B][1024][1024], Bt=V channel-major [B][512][1024].
// ---------------------------------------------------------------------------
__global__ __launch_bounds__(256)
void attn_gemm(const __bf16* __restrict__ Apm,
               const __bf16* __restrict__ Bt,
               int Kdim, int Ncols, float out_scale,
               float*  __restrict__ out_f32,
               __bf16* __restrict__ out_pm)
{
  __shared__ __align__(16) __bf16 Asub[2][128][32];
  __shared__ __align__(16) __bf16 Bsub[2][128][32];

  const int b    = blockIdx.z;
  const int m0   = blockIdx.y * 128;
  const int n0   = blockIdx.x * 128;
  const int tid  = threadIdx.x;
  const int wv   = tid >> 5;
  const int lane = tid & 31;
  const int hf   = lane >> 4;
  const int lm   = lane & 15;

  const v8f vzero = {0.f,0.f,0.f,0.f,0.f,0.f,0.f,0.f};
  v8f acc[8];
#pragma unroll
  for (int i = 0; i < 8; ++i) acc[i] = vzero;

  const int srow = tid >> 1;
  const int kh   = (tid & 1) * 16;
  const int nsteps = Kdim / 32;

  auto stage = [&](int step, int buf) {
    const int kc = step * 32;
    cp32(&Asub[buf][srow][kh],
         Apm + ((size_t)b * HWP + m0 + srow) * (size_t)Kdim + kc + kh);
    cp32(&Bsub[buf][srow][kh],
         Bt + ((size_t)b * Ncols + n0 + srow) * (size_t)Kdim + kc + kh);
  };

  stage(0, 0);
  for (int s = 0; s < nsteps; ++s) {
    const int cur = s & 1;
    if (s + 1 < nsteps) { stage(s + 1, cur ^ 1); wait_async_le4(); }
    else                { wait_async_le0(); }
    __syncthreads();

    Frag16 af;
    const int arow = wv * 16 + lm;
#pragma unroll
    for (int v = 0; v < 8; ++v) {
      const int kb = hf * 16 + v * 2;
      af.u[v] = *reinterpret_cast<const unsigned*>(&Asub[cur][arow][kb]);
    }
#pragma unroll
    for (int nt = 0; nt < 8; ++nt) {
      Frag16 bfr;
      const int brow = nt * 16 + lm;
#pragma unroll
      for (int v = 0; v < 8; ++v)
        bfr.u[v] = *reinterpret_cast<const unsigned*>(&Bsub[cur][brow][hf * 16 + v * 2]);
      acc[nt] = __builtin_amdgcn_wmma_f32_16x16x32_bf16(
          false, af.v, false, bfr.v, (short)0, acc[nt], false, false);
    }
    __syncthreads();
  }

#pragma unroll
  for (int nt = 0; nt < 8; ++nt) {
    const int col = n0 + nt * 16 + lm;
#pragma unroll
    for (int v = 0; v < 8; ++v) {
      const int row = m0 + wv * 16 + hf * 8 + v;
      const float val = acc[nt][v] * out_scale;
      const size_t o = ((size_t)b * HWP + row) * (size_t)Ncols + col;
      if (out_f32) out_f32[o] = val;
      if (out_pm)  out_pm[o]  = (__bf16)val;
    }
  }
}

// ---------------------------------------------------------------------------
// Fused GroupNorm (+optional SiLU, +optional residual add); dual-layout out.
// One workgroup per (batch, group): 16 contiguous channels = 16384 f32.
// ---------------------------------------------------------------------------
__global__ __launch_bounds__(256)
void k_gn(const float* __restrict__ in_nchw,
          const float* __restrict__ gamma, const float* __restrict__ beta,
          const float* __restrict__ resid, int do_silu,
          float* __restrict__ out_f32, __bf16* __restrict__ out_pm)
{
  __shared__ float s_sum[256];
  __shared__ float s_sq[256];
  __shared__ float s_stats[2];
  const int b = blockIdx.x >> 5;
  const int g = blockIdx.x & 31;
  const size_t base = ((size_t)b * CCH + g * CPG) * HWP;
  const int tid = threadIdx.x;

  float sm = 0.f, sq = 0.f;
  for (int e = tid; e < CPG * HWP; e += 256) {
    const float v = in_nchw[base + e];
    sm += v; sq += v * v;
  }
  s_sum[tid] = sm; s_sq[tid] = sq;
  __syncthreads();
  for (int off = 128; off > 0; off >>= 1) {
    if (tid < off) { s_sum[tid] += s_sum[tid + off]; s_sq[tid] += s_sq[tid + off]; }
    __syncthreads();
  }
  if (tid == 0) {
    const float inv_n = 1.0f / (float)(CPG * HWP);
    const float mean = s_sum[0] * inv_n;
    const float var  = s_sq[0] * inv_n - mean * mean;   // biased, matches jnp.var
    s_stats[0] = mean;
    s_stats[1] = rsqrtf(var + 1e-6f);
  }
  __syncthreads();
  const float mean = s_stats[0], rstd = s_stats[1];

  for (int e = tid; e < CPG * HWP; e += 256) {
    const int c = g * CPG + (e >> 10);
    const int p = e & 1023;
    float v = (in_nchw[base + e] - mean) * rstd * gamma[c] + beta[c];
    if (do_silu) v = v / (1.0f + expf(-v));
    if (resid)   v += resid[base + e];
    if (out_f32) out_f32[base + e] = v;
    if (out_pm)  out_pm[((size_t)b * HWP + p) * CCH + c] = (__bf16)v;
  }
}

// ---------------------------------------------------------------------------
// Row softmax: S f32 [B*HW][1024] -> P bf16, one workgroup per row.
// ---------------------------------------------------------------------------
__global__ __launch_bounds__(256)
void k_softmax(const float* __restrict__ S, __bf16* __restrict__ P)
{
  __shared__ float red[256];
  const size_t base = (size_t)blockIdx.x * 1024;
  const int tid = threadIdx.x;
  const float x0 = S[base + tid],       x1 = S[base + tid + 256];
  const float x2 = S[base + tid + 512], x3 = S[base + tid + 768];
  float mx = fmaxf(fmaxf(x0, x1), fmaxf(x2, x3));
  red[tid] = mx; __syncthreads();
  for (int off = 128; off > 0; off >>= 1) {
    if (tid < off) red[tid] = fmaxf(red[tid], red[tid + off]);
    __syncthreads();
  }
  mx = red[0]; __syncthreads();
  const float e0 = expf(x0 - mx), e1 = expf(x1 - mx);
  const float e2 = expf(x2 - mx), e3 = expf(x3 - mx);
  red[tid] = e0 + e1 + e2 + e3; __syncthreads();
  for (int off = 128; off > 0; off >>= 1) {
    if (tid < off) red[tid] += red[tid + off];
    __syncthreads();
  }
  const float inv = 1.0f / red[0];
  P[base + tid]       = (__bf16)(e0 * inv);
  P[base + tid + 256] = (__bf16)(e1 * inv);
  P[base + tid + 512] = (__bf16)(e2 * inv);
  P[base + tid + 768] = (__bf16)(e3 * inv);
}

// ---------------------------------------------------------------------------
// Host-side orchestration
// ---------------------------------------------------------------------------
extern "C" void kernel_launch(void* const* d_in, const int* in_sizes, int n_in,
                              void* d_out, int out_size, void* d_ws, size_t ws_size,
                              hipStream_t stream) {
  (void)in_sizes; (void)n_in; (void)out_size; (void)ws_size;

  const float* x      = (const float*)d_in[0];
  const float* r1c1w  = (const float*)d_in[1];
  const float* r1c1b  = (const float*)d_in[2];
  const float* r1g1   = (const float*)d_in[3];
  const float* r1b1   = (const float*)d_in[4];
  const float* r1c2w  = (const float*)d_in[5];
  const float* r1c2b  = (const float*)d_in[6];
  const float* r1g2   = (const float*)d_in[7];
  const float* r1b2   = (const float*)d_in[8];
  const float* ag     = (const float*)d_in[9];
  const float* ab     = (const float*)d_in[10];
  const float* aqw    = (const float*)d_in[11];
  const float* aqb    = (const float*)d_in[12];
  const float* akw    = (const float*)d_in[13];
  const float* akb    = (const float*)d_in[14];
  const float* avw    = (const float*)d_in[15];
  const float* avb    = (const float*)d_in[16];
  const float* apw    = (const float*)d_in[17];
  const float* apb    = (const float*)d_in[18];
  const float* r2c1w  = (const float*)d_in[19];
  const float* r2c1b  = (const float*)d_in[20];
  const float* r2g1   = (const float*)d_in[21];
  const float* r2b1   = (const float*)d_in[22];
  const float* r2c2w  = (const float*)d_in[23];
  const float* r2c2b  = (const float*)d_in[24];
  const float* r2g2   = (const float*)d_in[25];
  const float* r2b2   = (const float*)d_in[26];

  // Workspace carve-up
  char* ws = (char*)d_ws;
  size_t off = 0;
  auto alloc = [&](size_t bytes) -> void* {
    void* p = ws + off;
    off += (bytes + 255) & ~(size_t)255;
    return p;
  };
  const size_t W3B = (size_t)9 * CCH * CCH * sizeof(__bf16);
  const size_t W1B = (size_t)CCH * CCH * sizeof(__bf16);
  const size_t PMB = (size_t)BB * HWP * CCH * sizeof(__bf16);
  const size_t F32 = (size_t)BB * CCH * HWP * sizeof(float);

  __bf16* w3a = (__bf16*)alloc(W3B);   // r1 conv1
  __bf16* w3b = (__bf16*)alloc(W3B);   // r1 conv2
  __bf16* w3c = (__bf16*)alloc(W3B);   // r2 conv1
  __bf16* w3d = (__bf16*)alloc(W3B);   // r2 conv2
  __bf16* w1q = (__bf16*)alloc(W1B);
  __bf16* w1k = (__bf16*)alloc(W1B);
  __bf16* w1v = (__bf16*)alloc(W1B);
  __bf16* w1p = (__bf16*)alloc(W1B);
  __bf16* bfA = (__bf16*)alloc(PMB);
  __bf16* bfB = (__bf16*)alloc(PMB);
  __bf16* Qp  = (__bf16*)alloc(PMB);   // Q pixel-major  [b][n][c]
  __bf16* Kp  = (__bf16*)alloc(PMB);   // K pixel-major  [b][m][c]
  __bf16* Vt  = (__bf16*)alloc(PMB);   // V channel-major [b][c][m]
  float*  Smx = (float*)alloc((size_t)BB * HWP * HWP * sizeof(float));
  __bf16* Pmx = (__bf16*)alloc((size_t)BB * HWP * HWP * sizeof(__bf16));
  float*  t1  = (float*)alloc(F32);
  float*  t2  = (float*)alloc(F32);
  float*  x1  = (float*)alloc(F32);
  float*  x2  = (float*)alloc(F32);
  float*  outF = (float*)d_out;

  const dim3 blk(256);
  const dim3 gConv(HWP / 128, CCH / 128, BB);    // (8,4,16)
  const dim3 gScore(HWP / 128, HWP / 128, BB);   // (8,8,16)
  const dim3 gPV(CCH / 128, HWP / 128, BB);      // (4,8,16)
  const int  gGN = BB * 32;                      // 512 groups
  const int  gSM = BB * HWP;                     // 16384 rows
  const float qscale = 1.0f / sqrtf((float)CCH);

  // Layout prep
  k_w3_bf16<<<(9 * CCH * CCH + 255) / 256, blk, 0, stream>>>(r1c1w, w3a);
  k_w3_bf16<<<(9 * CCH * CCH + 255) / 256, blk, 0, stream>>>(r1c2w, w3b);
  k_w3_bf16<<<(9 * CCH * CCH + 255) / 256, blk, 0, stream>>>(r2c1w, w3c);
  k_w3_bf16<<<(9 * CCH * CCH + 255) / 256, blk, 0, stream>>>(r2c2w, w3d);
  k_w1_bf16<<<(CCH * CCH + 255) / 256, blk, 0, stream>>>(aqw, w1q);
  k_w1_bf16<<<(CCH * CCH + 255) / 256, blk, 0, stream>>>(akw, w1k);
  k_w1_bf16<<<(CCH * CCH + 255) / 256, blk, 0, stream>>>(avw, w1v);
  k_w1_bf16<<<(CCH * CCH + 255) / 256, blk, 0, stream>>>(apw, w1p);
  k_x_pm<<<(BB * CCH * HWP + 255) / 256, blk, 0, stream>>>(x, bfA);

  // ResNet block 1
  conv_gemm<<<gConv, blk, 0, stream>>>(bfA, w3a, r1c1b, 9, 1.0f, nullptr, t1, nullptr, nullptr);
  k_gn<<<gGN, blk, 0, stream>>>(t1, r1g1, r1b1, nullptr, 1, nullptr, bfB);
  conv_gemm<<<gConv, blk, 0, stream>>>(bfB, w3b, r1c2b, 9, 1.0f, nullptr, t2, nullptr, nullptr);
  k_gn<<<gGN, blk, 0, stream>>>(t2, r1g2, r1b2, x, 1, x1, nullptr);      // x1 = silu(gn(h2)) + x

  // Attention block
  k_gn<<<gGN, blk, 0, stream>>>(x1, ag, ab, nullptr, 0, nullptr, bfA);   // hn (bf16 pm)
  conv_gemm<<<gConv, blk, 0, stream>>>(bfA, w1q, aqb, 1, qscale, nullptr, nullptr, Qp, nullptr);
  conv_gemm<<<gConv, blk, 0, stream>>>(bfA, w1k, akb, 1, 1.0f, nullptr, nullptr, Kp, nullptr);
  conv_gemm<<<gConv, blk, 0, stream>>>(bfA, w1v, avb, 1, 1.0f, nullptr, nullptr, nullptr, Vt);
  attn_gemm<<<gScore, blk, 0, stream>>>(Qp, Kp, CCH, HWP, 1.0f, Smx, nullptr);   // S = Q K^T
  k_softmax<<<gSM, blk, 0, stream>>>(Smx, Pmx);
  attn_gemm<<<gPV, blk, 0, stream>>>(Pmx, Vt, HWP, CCH, 1.0f, nullptr, bfB);     // H = P V
  conv_gemm<<<gConv, blk, 0, stream>>>(bfB, w1p, apb, 1, 1.0f, x1, x2, bfA, nullptr); // x2 = x1 + proj(H)

  // ResNet block 2
  conv_gemm<<<gConv, blk, 0, stream>>>(bfA, w3c, r2c1b, 9, 1.0f, nullptr, t1, nullptr, nullptr);
  k_gn<<<gGN, blk, 0, stream>>>(t1, r2g1, r2b1, nullptr, 1, nullptr, bfB);
  conv_gemm<<<gConv, blk, 0, stream>>>(bfB, w3d, r2c2b, 9, 1.0f, nullptr, t2, nullptr, nullptr);
  k_gn<<<gGN, blk, 0, stream>>>(t2, r2g2, r2b2, x2, 1, outF, nullptr);   // final output
}
